// NONLocalBlock1D_61564061221373
// MI455X (gfx1250) — matrix operand order repaired and checked by reading, not compiled, not run on hardware
//
#include <hip/hip_runtime.h>
#include <hip/hip_bf16.h>
#include <stdint.h>

// ---------------------------------------------------------------------------
// Problem constants (from reference)
// ---------------------------------------------------------------------------
#define B_   8
#define C_   256
#define IC_  128
#define T_   4096
#define S_   2048            // T/2 after maxpool
#define BN_EPS 1e-5f

// ---------------------------------------------------------------------------
// Types for CDNA5 WMMA (wave32, 16x16x32 bf16 -> f32)
// ---------------------------------------------------------------------------
typedef __attribute__((ext_vector_type(16))) __bf16 v16bf;
typedef __attribute__((ext_vector_type(8)))  __bf16 v8bf;
typedef __attribute__((ext_vector_type(8)))  float  v8f;

union BF16Frag {
    v8bf  h[2];
    v16bf v;
};

#define DEVI __device__ __forceinline__

DEVI __bf16 f2bf(float f) { return (__bf16)f; }   // native trunc (RNE)

DEVI v8f wmma_bf16(v16bf a, v16bf b, v8f c) {
    // D = A(16x32 bf16) * B(32x16 bf16) + C(16x16 f32)
    return __builtin_amdgcn_wmma_f32_16x16x32_bf16(
        /*neg_a=*/false, a, /*neg_b=*/false, b,
        /*c_mod=*/(short)0, c, /*reuse_a=*/false, /*reuse_b=*/false);
}

// CDNA5 async global->LDS copy (16B per lane), tracked by ASYNCcnt.
DEVI void async_load_b128(unsigned int lds_off, const void* gptr) {
    asm volatile("global_load_async_to_lds_b128 %0, %1, off"
                 :: "v"(lds_off), "v"(gptr) : "memory");
}
DEVI void wait_asynccnt_4() { asm volatile("s_wait_asynccnt 0x4" ::: "memory"); }
DEVI void wait_asynccnt_0() { asm volatile("s_wait_asynccnt 0x0" ::: "memory"); }

// ---------------------------------------------------------------------------
// Kernel 0: one-shot fp32 -> bf16 conversion of all four weight matrices.
// ---------------------------------------------------------------------------
__global__ __launch_bounds__(256)
void prep_weights_kernel(const float* __restrict__ tw, const float* __restrict__ pw,
                         const float* __restrict__ gw, const float* __restrict__ ww,
                         __bf16* __restrict__ otw, __bf16* __restrict__ opw,
                         __bf16* __restrict__ ogw, __bf16* __restrict__ oww)
{
    const int i = blockIdx.x * 256 + threadIdx.x;
    if (i < IC_ * C_) {
        otw[i] = f2bf(tw[i]);
        opw[i] = f2bf(pw[i]);
        ogw[i] = f2bf(gw[i]);
        oww[i] = f2bf(ww[i]);
    }
}

// ---------------------------------------------------------------------------
// Kernel 1: front convs (theta / phi / g), phi & g maxpooled by 2.
//   theta_t : bf16 (B, T, IC)  -- A-operand rows for the f matmul
//   phi_t   : bf16 (B, S, IC)  -- K=ic contiguous: B-operand for f matmul
//   g       : bf16 (B, IC, S)  -- K=s contiguous: B-operand for y matmul
// Block: 128 threads (4 waves), each block handles (b, 64 t-positions).
// ---------------------------------------------------------------------------
template <int CV>
DEVI void front_conv(const __bf16* __restrict__ W, const float* __restrict__ Bb,
                     const BF16Frag (&xf)[8],
                     int b, int t0b, int tcol, int lane, int nrow, int khalf,
                     __bf16* __restrict__ theta_t, __bf16* __restrict__ phi_t,
                     __bf16* __restrict__ g)
{
#pragma unroll
    for (int m = 0; m < 8; ++m) {                 // IC tiles of 16
        v8f acc = {};
        const int rowic = m * 16 + nrow;          // A row (output channel)
#pragma unroll
        for (int kc = 0; kc < 8; ++kc) {          // C in chunks of 32
            const __bf16* wp = W + (size_t)rowic * C_ + kc * 32 + khalf * 8;
            BF16Frag af;
            af.h[0] = *(const v8bf*)wp;
            af.h[1] = *(const v8bf*)(wp + 16);
            acc = wmma_bf16(af.v, xf[kc].v, acc);
        }
        const int chb = m * 16 + 8 * khalf;       // first channel of this lane's 8 D rows
#pragma unroll
        for (int r = 0; r < 8; ++r) acc[r] += Bb[chb + r];

        if (CV == 0) {
            // theta: (B, T, IC); lane owns fixed t, 8 contiguous ic -> one b128 store
            v8bf pk;
#pragma unroll
            for (int r = 0; r < 8; ++r) pk[r] = f2bf(acc[r]);
            *(v8bf*)&theta_t[((size_t)b * T_ + (t0b + tcol)) * IC_ + chb] = pk;
        } else {
            // maxpool over adjacent t (adjacent lanes within each 16-lane half)
            float pooled[8];
#pragma unroll
            for (int r = 0; r < 8; ++r) {
                float v = acc[r];
                pooled[r] = fmaxf(v, __shfl_xor(v, 1, 32));
            }
            if ((lane & 1) == 0) {
                const int s = (t0b + tcol) >> 1;
                if (CV == 1) {
                    // phi_t: (B, S, IC); fixed s, 8 contiguous ic -> one b128 store
                    v8bf pk;
#pragma unroll
                    for (int r = 0; r < 8; ++r) pk[r] = f2bf(pooled[r]);
                    *(v8bf*)&phi_t[((size_t)b * S_ + s) * IC_ + chb] = pk;
                } else {
                    // g: (B, IC, S); channel-strided scalar stores
#pragma unroll
                    for (int r = 0; r < 8; ++r)
                        g[((size_t)b * IC_ + chb + r) * S_ + s] = f2bf(pooled[r]);
                }
            }
        }
    }
}

__global__ __launch_bounds__(128)
void conv_front_kernel(const float* __restrict__ x,
                       const __bf16* __restrict__ tw, const float* __restrict__ tb,
                       const __bf16* __restrict__ pw, const float* __restrict__ pb,
                       const __bf16* __restrict__ gw, const float* __restrict__ gb,
                       __bf16* __restrict__ theta_t,
                       __bf16* __restrict__ phi_t,
                       __bf16* __restrict__ g)
{
    __shared__ __align__(16) __bf16 x_lds[64][264];  // [t][c], stride 528B (16B mult)

    const int b    = blockIdx.x / (T_ / 64);
    const int tile = blockIdx.x % (T_ / 64);
    const int t0b  = tile * 64;

    // Stage x[b][:, t0b..t0b+63] fp32 -> bf16 transposed to [t][c]:
    // each thread gathers 8 channel-strided values (coalesced along t across
    // threads) and writes one packed ds_store_b128.
    {
        const int tloc = threadIdx.x & 63;
        const int cgrp = threadIdx.x >> 6;       // 0..1
#pragma unroll
        for (int pass = 0; pass < 16; ++pass) {
            const int c0 = pass * 16 + cgrp * 8;
            v8bf pk;
#pragma unroll
            for (int k = 0; k < 8; ++k)
                pk[k] = f2bf(x[((size_t)b * C_ + c0 + k) * T_ + t0b + tloc]);
            *(v8bf*)&x_lds[tloc][c0] = pk;
        }
    }
    __syncthreads();

    const int lane  = threadIdx.x & 31;
    const int w     = threadIdx.x >> 5;
    const int nrow  = lane & 15;
    const int khalf = lane >> 4;
    const int tcol  = w * 16 + nrow;

    // x B-fragments for all 8 K-chunks (B[K=c][N=t]).
    BF16Frag xf[8];
#pragma unroll
    for (int kc = 0; kc < 8; ++kc) {
        const __bf16* p = &x_lds[tcol][kc * 32 + khalf * 8];
        xf[kc].h[0] = *(const v8bf*)p;
        xf[kc].h[1] = *(const v8bf*)(p + 16);
    }

    front_conv<0>(tw, tb, xf, b, t0b, tcol, lane, nrow, khalf, theta_t, phi_t, g);
    front_conv<1>(pw, pb, xf, b, t0b, tcol, lane, nrow, khalf, theta_t, phi_t, g);
    front_conv<2>(gw, gb, xf, b, t0b, tcol, lane, nrow, khalf, theta_t, phi_t, g);
}

// ---------------------------------------------------------------------------
// Kernel 2: fused attention  y = (theta^T phi / S) g^T  (f never materialized)
//   theta_t (B,T,IC), phi_t (B,S,IC), g (B,IC,S)  all bf16 -> y bf16 (B,IC,T)
// Block: 256 threads (8 waves), each wave owns a 16-row t-tile; s streamed in
// chunks of 32 through double-buffered LDS filled with async b128 copies.
// ---------------------------------------------------------------------------
__global__ __launch_bounds__(256)
void attention_kernel(const __bf16* __restrict__ theta_t,
                      const __bf16* __restrict__ phi_t,
                      const __bf16* __restrict__ g,
                      __bf16* __restrict__ y)
{
    __shared__ __align__(16) __bf16 phi_lds[2][32][136];   // [buf][s][ic], row 272B
    __shared__ __align__(16) __bf16 g_lds[2][128][40];     // [buf][ic][s], row 80B
    __shared__ __align__(16) __bf16 f_lds[8][16][40];      // per-wave f restripe

    const int b    = blockIdx.x / (T_ / 128);
    const int tile = blockIdx.x % (T_ / 128);
    const int t0b  = tile * 128;

    const int tid   = threadIdx.x;
    const int lane  = tid & 31;
    const int w     = tid >> 5;
    const int nrow  = lane & 15;
    const int khalf = lane >> 4;
    const int t0    = t0b + w * 16;

    const unsigned int phi_off0 = (unsigned int)(uintptr_t)&phi_lds[0][0][0];
    const unsigned int g_off0   = (unsigned int)(uintptr_t)&g_lds[0][0][0];
    const unsigned int phi_bufsz = 32 * 136 * 2;
    const unsigned int g_bufsz   = 128 * 40 * 2;

    // theta A-fragments (K = IC = 128 -> 4 chunks), resident for whole kernel.
    BF16Frag th[4];
    {
        const __bf16* tr = theta_t + ((size_t)b * T_ + t0 + nrow) * IC_;
#pragma unroll
        for (int kc = 0; kc < 4; ++kc) {
            th[kc].h[0] = *(const v8bf*)&tr[kc * 32 + khalf * 8];
            th[kc].h[1] = *(const v8bf*)&tr[kc * 32 + 16 + khalf * 8];
        }
    }

    v8f yacc[8];
    {
        v8f z = {};
#pragma unroll
        for (int n = 0; n < 8; ++n) yacc[n] = z;
    }

    const float inv_n = 1.0f / (float)S_;

    // async staging of one 32-wide s-chunk (4 x b128 per thread)
    auto stage_chunk = [&](int sc, int buf) {
        // phi_t rows: 32 rows x 256B  -> 512 x 16B segments
        {
            const int seg  = tid & 15;
            const int row0 = tid >> 4;                    // 0..15
#pragma unroll
            for (int k = 0; k < 2; ++k) {
                const int row = row0 + 16 * k;
                const __bf16* gp = phi_t + ((size_t)b * S_ + sc + row) * IC_ + seg * 8;
                async_load_b128(phi_off0 + buf * phi_bufsz + row * 272 + seg * 16, gp);
            }
        }
        // g rows: 128 rows x 64B -> 512 x 16B segments
        {
            const int row = tid >> 1;                     // 0..127
#pragma unroll
            for (int k = 0; k < 2; ++k) {
                const int seg = (tid & 1) + 2 * k;
                const __bf16* gp = g + ((size_t)b * IC_ + row) * S_ + sc + seg * 8;
                async_load_b128(g_off0 + buf * g_bufsz + row * 80 + seg * 16, gp);
            }
        }
    };

    const int NIT = S_ / 32;                              // 64
    stage_chunk(0, 0);

    for (int it = 0; it < NIT; ++it) {
        const int buf = it & 1;
        if (it + 1 < NIT) {
            stage_chunk((it + 1) * 32, buf ^ 1);
            wait_asynccnt_4();          // chunk `it` (issued earlier) has landed
        } else {
            wait_asynccnt_0();
        }
        __syncthreads();

        // f tile: 16 t x 32 s, two 16x16 WMMA outputs, K = IC = 128 (4 chunks)
#pragma unroll
        for (int h = 0; h < 2; ++h) {
            v8f facc = {};
            const int scol = h * 16 + nrow;
#pragma unroll
            for (int kc = 0; kc < 4; ++kc) {
                const __bf16* p = &phi_lds[buf][scol][kc * 32 + khalf * 8];
                BF16Frag bfr;
                bfr.h[0] = *(const v8bf*)p;
                bfr.h[1] = *(const v8bf*)(p + 16);
                facc = wmma_bf16(th[kc].v, bfr.v, facc);
            }
            // restripe D-layout -> A-layout via per-wave LDS scratch (scaled 1/S)
#pragma unroll
            for (int r = 0; r < 8; ++r)
                f_lds[w][r + 8 * khalf][h * 16 + nrow] = f2bf(facc[r] * inv_n);
        }
        // same-wave LDS RAW: DS ops are in-order within a wave
        BF16Frag ff;
        ff.h[0] = *(const v8bf*)&f_lds[w][nrow][khalf * 8];
        ff.h[1] = *(const v8bf*)&f_lds[w][nrow][16 + khalf * 8];

        // y(tile) += f(16x32) * g^T(32x128): 8 N-tiles of 16 ic
#pragma unroll
        for (int n = 0; n < 8; ++n) {
            const __bf16* gp = &g_lds[buf][n * 16 + nrow][khalf * 8];
            BF16Frag gf;
            gf.h[0] = *(const v8bf*)gp;
            gf.h[1] = *(const v8bf*)(gp + 16);
            yacc[n] = wmma_bf16(ff.v, gf.v, yacc[n]);
        }
        __syncthreads();   // protect buf from next iteration's async fill
    }

    // store y (B, IC, T): lane owns fixed ic, 8 contiguous t
#pragma unroll
    for (int n = 0; n < 8; ++n) {
        v8bf pk;
#pragma unroll
        for (int r = 0; r < 8; ++r) pk[r] = f2bf(yacc[n][r]);
        const int ic = n * 16 + nrow;
        const int tb = t0 + 8 * khalf;
        *(v8bf*)&y[((size_t)b * IC_ + ic) * T_ + tb] = pk;
    }
}

// ---------------------------------------------------------------------------
// Kernel 3: back conv  w_y = w_w @ y + w_b   (fp32 output, for BN accuracy)
// ---------------------------------------------------------------------------
__global__ __launch_bounds__(128)
void conv_back_kernel(const __bf16* __restrict__ y,
                      const __bf16* __restrict__ ww, const float* __restrict__ wb,
                      float* __restrict__ w_y)
{
    __shared__ __align__(16) __bf16 y_lds[64][136];   // [t][ic], row 272B

    const int b    = blockIdx.x / (T_ / 64);
    const int tile = blockIdx.x % (T_ / 64);
    const int t0b  = tile * 64;

    // stage y[b][:, t0b..t0b+63] transposed to [t][ic]: gather 8 channel-
    // strided bf16 (coalesced along t) -> one packed ds_store_b128.
    {
        const int tloc = threadIdx.x & 63;
        const int grp  = threadIdx.x >> 6;       // 0..1
#pragma unroll
        for (int pass = 0; pass < 8; ++pass) {
            const int ic0 = pass * 16 + grp * 8;
            v8bf pk;
#pragma unroll
            for (int k = 0; k < 8; ++k)
                pk[k] = y[((size_t)b * IC_ + ic0 + k) * T_ + t0b + tloc];
            *(v8bf*)&y_lds[tloc][ic0] = pk;
        }
    }
    __syncthreads();

    const int lane  = threadIdx.x & 31;
    const int w     = threadIdx.x >> 5;
    const int nrow  = lane & 15;
    const int khalf = lane >> 4;
    const int tcol  = w * 16 + nrow;

    BF16Frag yf[4];
#pragma unroll
    for (int kc = 0; kc < 4; ++kc) {
        const __bf16* p = &y_lds[tcol][kc * 32 + khalf * 8];
        yf[kc].h[0] = *(const v8bf*)p;
        yf[kc].h[1] = *(const v8bf*)(p + 16);
    }

#pragma unroll
    for (int m = 0; m < 16; ++m) {               // C = 256 -> 16 M-tiles
        v8f acc = {};
        const int rowc = m * 16 + nrow;
#pragma unroll
        for (int kc = 0; kc < 4; ++kc) {
            const __bf16* wp = ww + (size_t)rowc * IC_ + kc * 32 + khalf * 8;
            BF16Frag af;
            af.h[0] = *(const v8bf*)wp;
            af.h[1] = *(const v8bf*)(wp + 16);
            acc = wmma_bf16(af.v, yf[kc].v, acc);
        }
        const int chb = m * 16 + 8 * khalf;
        const int t   = t0b + tcol;
#pragma unroll
        for (int r = 0; r < 8; ++r)
            w_y[((size_t)b * C_ + chb + r) * T_ + t] = acc[r] + wb[chb + r];
    }
}

// ---------------------------------------------------------------------------
// Kernel 4: per-channel BatchNorm statistics over (B, T)
// ---------------------------------------------------------------------------
__global__ __launch_bounds__(256)
void bn_stats_kernel(const float* __restrict__ w_y, float* __restrict__ stats)
{
    const int c = blockIdx.x;
    float s = 0.f, ss = 0.f;
    for (int b = 0; b < B_; ++b) {
        const float* p = w_y + ((size_t)b * C_ + c) * T_;
        for (int t = threadIdx.x; t < T_; t += 256) {
            float v = p[t];
            s  += v;
            ss += v * v;
        }
    }
    __shared__ float rs[256], rss[256];
    rs[threadIdx.x]  = s;
    rss[threadIdx.x] = ss;
    __syncthreads();
    for (int off = 128; off > 0; off >>= 1) {
        if (threadIdx.x < off) {
            rs[threadIdx.x]  += rs[threadIdx.x + off];
            rss[threadIdx.x] += rss[threadIdx.x + off];
        }
        __syncthreads();
    }
    if (threadIdx.x == 0) {
        const float n    = (float)(B_ * T_);
        const float mean = rs[0] / n;
        const float var  = rss[0] / n - mean * mean;
        stats[c]        = mean;
        stats[C_ + c]   = var;
    }
}

// ---------------------------------------------------------------------------
// Kernel 5: BN affine + residual
// ---------------------------------------------------------------------------
__global__ __launch_bounds__(256)
void finalize_kernel(const float* __restrict__ w_y, const float* __restrict__ x,
                     const float* __restrict__ stats,
                     const float* __restrict__ gamma, const float* __restrict__ beta,
                     float* __restrict__ out)
{
    const size_t total = (size_t)B_ * C_ * T_;
    for (size_t idx = (size_t)blockIdx.x * blockDim.x + threadIdx.x;
         idx < total; idx += (size_t)gridDim.x * blockDim.x) {
        const int c = (int)((idx / T_) % C_);
        const float mean = stats[c];
        const float var  = stats[C_ + c];
        out[idx] = (w_y[idx] - mean) * rsqrtf(var + BN_EPS) * gamma[c] + beta[c] + x[idx];
    }
}

// ---------------------------------------------------------------------------
// Launch
// ---------------------------------------------------------------------------
extern "C" void kernel_launch(void* const* d_in, const int* in_sizes, int n_in,
                              void* d_out, int out_size, void* d_ws, size_t ws_size,
                              hipStream_t stream)
{
    const float* x        = (const float*)d_in[0];
    const float* theta_w  = (const float*)d_in[1];
    const float* theta_b  = (const float*)d_in[2];
    const float* phi_w    = (const float*)d_in[3];
    const float* phi_b    = (const float*)d_in[4];
    const float* g_w      = (const float*)d_in[5];
    const float* g_b      = (const float*)d_in[6];
    const float* w_w      = (const float*)d_in[7];
    const float* w_b      = (const float*)d_in[8];
    const float* bn_gamma = (const float*)d_in[9];
    const float* bn_beta  = (const float*)d_in[10];
    float* out = (float*)d_out;

    // workspace layout (bytes); w_y reuses the theta/phi/g region (dead after attention)
    char* ws = (char*)d_ws;
    const size_t y_bytes     = (size_t)B_ * IC_ * T_ * 2;   //  8 MB bf16
    const size_t theta_bytes = (size_t)B_ * T_ * IC_ * 2;   //  8 MB bf16
    const size_t phi_bytes   = (size_t)B_ * S_ * IC_ * 2;   //  4 MB bf16
    const size_t wy_bytes    = (size_t)B_ * C_ * T_ * 4;    // 32 MB fp32
    const size_t wmat_bytes  = (size_t)IC_ * C_ * 2;        // 64 KB bf16 each

    __bf16* y_buf   = (__bf16*)(ws);
    __bf16* theta_t = (__bf16*)(ws + y_bytes);
    __bf16* phi_t   = (__bf16*)(ws + y_bytes + theta_bytes);
    __bf16* g_buf   = (__bf16*)(ws + y_bytes + theta_bytes + phi_bytes);
    float*  w_y     = (float*)(ws + y_bytes);               // overlaps theta/phi/g (safe)
    float*  stats   = (float*)(ws + y_bytes + wy_bytes);
    char*   wbase   = ws + y_bytes + wy_bytes + 4096;       // persistent bf16 weights
    __bf16* tw_bf   = (__bf16*)(wbase);
    __bf16* pw_bf   = (__bf16*)(wbase + wmat_bytes);
    __bf16* gw_bf   = (__bf16*)(wbase + 2 * wmat_bytes);
    __bf16* ww_bf   = (__bf16*)(wbase + 3 * wmat_bytes);
    (void)in_sizes; (void)n_in; (void)out_size; (void)ws_size;

    prep_weights_kernel<<<(IC_ * C_ + 255) / 256, 256, 0, stream>>>(
        theta_w, phi_w, g_w, w_w, tw_bf, pw_bf, gw_bf, ww_bf);

    conv_front_kernel<<<B_ * (T_ / 64), 128, 0, stream>>>(
        x, tw_bf, theta_b, pw_bf, phi_b, gw_bf, g_b, theta_t, phi_t, g_buf);

    attention_kernel<<<B_ * (T_ / 128), 256, 0, stream>>>(theta_t, phi_t, g_buf, y_buf);

    conv_back_kernel<<<B_ * (T_ / 64), 128, 0, stream>>>(y_buf, ww_bf, w_b, w_y);

    bn_stats_kernel<<<C_, 256, 0, stream>>>(w_y, stats);

    finalize_kernel<<<2048, 256, 0, stream>>>(w_y, x, stats, bn_gamma, bn_beta, out);
}
